// Renderer_69509750719074
// MI455X (gfx1250) — matrix operand order repaired and checked by reading, not compiled, not run on hardware
//
#include <hip/hip_runtime.h>

typedef __attribute__((ext_vector_type(16))) _Float16 v16h;
typedef __attribute__((ext_vector_type(8)))  float    v8f;
typedef __attribute__((ext_vector_type(4)))  float    f4;

#define GAIN_     2.0f
#define BASELINE_ 100.0f
#define B_  32
#define E_  64
#define H_  128
#define W_  128
#define R_  13
#define D_  32
#define RR64 (R_ * R_ * 64)   /* 10816 floats per z-slice */

__global__ __launch_bounds__(256) void smlm_render_kernel(
    const float* __restrict__ xyz,
    const float* __restrict__ nph,
    const float* __restrict__ bg,
    const float* __restrict__ coeff,
    float* __restrict__ out)
{
    __shared__ float patch_s[176];          // 169 used, padded to 11 tiles of 16

    const int be  = blockIdx.x;             // b*E + e
    const int b   = be >> 6;                // /E_
    const int tid = threadIdx.x;

    // Per-emitter scalars (all threads; needed by the streaming phase too)
    const float x = xyz[be * 3 + 0];
    const float y = xyz[be * 3 + 1];
    const float z = xyz[be * 3 + 2];
    const float xf = floorf(x), yf = floorf(y);
    const int r0 = (int)yf - (R_ / 2);
    const int c0 = (int)xf - (R_ / 2);

    // ---- Phase 1: wave 0 evaluates the 13x13 patch via WMMA --------------
    if (tid < 32) {
        const int lane = tid;
        const float dx = x - xf, dy = y - yf;
        float zc = fminf(fmaxf(z, 0.0f), (float)(D_ - 1) - 1e-6f);
        const float zf = floorf(zc);
        const int   zi = (int)zf;
        const float dz = zc - zf;

        float px[4], py[4], pz[4];
        px[0] = 1.0f; px[1] = dx; px[2] = dx * dx; px[3] = px[2] * dx;
        py[0] = 1.0f; py[1] = dy; py[2] = dy * dy; py[3] = py[2] * dy;
        pz[0] = 1.0f; pz[1] = dz; pz[2] = dz * dz; pz[3] = pz[2] * dz;

        const float scale = nph[be] * GAIN_;
        const int h = lane >> 4;            // lane half (K-half selector)
        const int m = lane & 15;            // A-matrix row within tile
        const float* slice = coeff + (size_t)zi * RR64;

        // B matrix: w[k] = pz[a]*py[b]*px[c], k = a*16+b*4+c, replicated
        // across all N columns (depends only on lane half h) -> every D
        // column contains the same patch values.
        v16h bv0, bv1;
        #pragma unroll
        for (int j = 0; j < 16; ++j) {
            int k0 = h * 8 + j + (j & 8);   // chunk 0: k = 0..31
            int k1 = 32 + h * 8 + j + (j & 8); // chunk 1: k = 32..63
            bv0[j] = (_Float16)(pz[k0 >> 4] * py[(k0 >> 2) & 3] * px[k0 & 3]);
            bv1[j] = (_Float16)(pz[k1 >> 4] * py[(k1 >> 2) & 3] * px[k1 & 3]);
        }

        for (int tile = 0; tile < 11; ++tile) {
            v8f acc = {};
            const float* rowp = slice + (tile * 16 + m) * 64 + 8 * h;
            #pragma unroll
            for (int ch = 0; ch < 2; ++ch) {
                const float* ap = rowp + ch * 32;
                // A layout (16-bit 16x32): element j holds
                // k = ch*32 + (j<8 ? 0 : 16) + 8*h + (j&7)
                float4 f0 = *(const float4*)(ap);
                float4 f1 = *(const float4*)(ap + 4);
                float4 f2 = *(const float4*)(ap + 16);
                float4 f3 = *(const float4*)(ap + 20);
                v16h av;
                av[0]  = (_Float16)f0.x; av[1]  = (_Float16)f0.y;
                av[2]  = (_Float16)f0.z; av[3]  = (_Float16)f0.w;
                av[4]  = (_Float16)f1.x; av[5]  = (_Float16)f1.y;
                av[6]  = (_Float16)f1.z; av[7]  = (_Float16)f1.w;
                av[8]  = (_Float16)f2.x; av[9]  = (_Float16)f2.y;
                av[10] = (_Float16)f2.z; av[11] = (_Float16)f2.w;
                av[12] = (_Float16)f3.x; av[13] = (_Float16)f3.y;
                av[14] = (_Float16)f3.z; av[15] = (_Float16)f3.w;
                acc = __builtin_amdgcn_wmma_f32_16x16x32_f16(
                    false, av, false, (ch == 0 ? bv0 : bv1),
                    (short)0, acc, false, false);
            }
            // Column n==0 (lanes 0 and 16) holds the full 16-row result.
            if (m == 0) {
                const int mb = tile * 16 + 8 * h;
                patch_s[mb + 0] = acc[0] * scale;
                patch_s[mb + 1] = acc[1] * scale;
                patch_s[mb + 2] = acc[2] * scale;
                patch_s[mb + 3] = acc[3] * scale;
                patch_s[mb + 4] = acc[4] * scale;
                patch_s[mb + 5] = acc[5] * scale;
                patch_s[mb + 6] = acc[6] * scale;
                patch_s[mb + 7] = acc[7] * scale;
            }
        }
    }
    __syncthreads();

    // ---- Phase 2: all 8 waves stream the 128x128 image -------------------
    const f4* bg4  = (const f4*)(bg + (size_t)b * (H_ * W_));
    f4*       out4 = (f4*)(out + (size_t)be * (H_ * W_));
    #pragma unroll
    for (int it = 0; it < 16; ++it) {
        const int i4 = it * 256 + tid;      // float4 index, 0..4095
        f4 v = bg4[i4];                     // L2-resident (bg reused 64x)
        f4 o;
        o.x = fmaf(v.x, GAIN_, BASELINE_);
        o.y = fmaf(v.y, GAIN_, BASELINE_);
        o.z = fmaf(v.z, GAIN_, BASELINE_);
        o.w = fmaf(v.w, GAIN_, BASELINE_);
        const int row = i4 >> 5;            // 32 float4s per image row
        const int pr  = row - r0;
        if ((unsigned)pr < (unsigned)R_) {  // inside patch rows
            const int pc = ((i4 & 31) << 2) - c0;
            const float* prow = patch_s + pr * R_;
            if ((unsigned)(pc)     < (unsigned)R_) o.x += prow[pc];
            if ((unsigned)(pc + 1) < (unsigned)R_) o.y += prow[pc + 1];
            if ((unsigned)(pc + 2) < (unsigned)R_) o.z += prow[pc + 2];
            if ((unsigned)(pc + 3) < (unsigned)R_) o.w += prow[pc + 3];
        }
        // Non-temporal: 134 MB of output must not evict bg/coeff from L2.
        __builtin_nontemporal_store(o, &out4[i4]);
    }
}

extern "C" void kernel_launch(void* const* d_in, const int* in_sizes, int n_in,
                              void* d_out, int out_size, void* d_ws, size_t ws_size,
                              hipStream_t stream) {
    (void)in_sizes; (void)n_in; (void)d_ws; (void)ws_size; (void)out_size;
    const float* xyz   = (const float*)d_in[0];
    const float* nphot = (const float*)d_in[1];
    const float* bg    = (const float*)d_in[2];
    const float* coeff = (const float*)d_in[3];
    float* out = (float*)d_out;
    dim3 grid(B_ * E_), block(256);
    hipLaunchKernelGGL(smlm_render_kernel, grid, block, 0, stream,
                       xyz, nphot, bg, coeff, out);
}
